// SwinTransformerBlock_89275190215404
// MI455X (gfx1250) — compile-verified, matrix operand order checked by reading, not run on hardware
//
#include <hip/hip_runtime.h>
#include <hip/hip_fp16.h>

typedef __attribute__((ext_vector_type(16))) _Float16 v16h;
typedef __attribute__((ext_vector_type(8)))  float    v8f;
typedef int v4i __attribute__((ext_vector_type(4)));
typedef __attribute__((address_space(1))) v4i* glob_v4i;
typedef __attribute__((address_space(3))) v4i* lds_v4i;

#define TOKENS 64
#define DIM    256
#define HEADS  8
#define HD     32
#define FF     1024

// ---- workspace (d_ws) layout: f16 weights + expanded rel-pos bias ----
#define WQKV_OFF  0                              // 768x256 f16
#define WPROJ_OFF (WQKV_OFF  + 768*256*2)        // 256x256 f16
#define WFC1_OFF  (WPROJ_OFF + 256*256*2)        // 1024x256 f16
#define WFC2_OFF  (WFC1_OFF  + 1024*256*2)       // 256x1024 f16
#define BIAS_OFF  (WFC2_OFF  + 256*1024*2)       // 8x64x64 f32

// ---- dynamic LDS layout (288 KB; CDNA5 WGP allows 320 KB) ----
#define XS_OFF    0                              // 64x256 f32  residual
#define HA_OFF    (64*256*4)                     // 64x256 f16  ln1 / attn-out / ln2
#define QKV_OFF   (HA_OFF + 64*256*2)            // 64x768 f16  qkv, reused 64x1024 mlp hidden
#define PBUF_OFF  (QKV_OFF + 64*768*2)           // 8x64x64 f16 softmax probs
#define VT_OFF    (PBUF_OFF + 8*64*64*2)         // 256x64 f16  V transposed (B-operand of P@V)
#define LDS_BYTES (VT_OFF + 256*64*2)            // 294912

static __device__ inline v8f wmma32(v16h a, v16h b, v8f c) {
  // D = A(16x32 f16) x B(32x16 f16) + C(16x16 f32)
  return __builtin_amdgcn_wmma_f32_16x16x32_f16(false, a, false, b, (short)0, c,
                                                false, false);
}

// A-fragment (16x32, rows = M) or B-as-transposed (rows = N) from a row-major
// f16 matrix. Lane l holds row r0+(l&15); K-halves per ISA: lanes 0-15 get
// K = {0..7, 16..23}+0, lanes 16-31 get +8.  Two 16-byte loads per lane.
static __device__ inline v16h load_frag_row(const _Float16* __restrict__ p, int ld,
                                            int r0, int k0) {
  const int lane = threadIdx.x & 31;
  const int rr = r0 + (lane & 15);
  const int kb = (lane >> 4) << 3;
  const _Float16* base = p + (size_t)rr * ld + k0 + kb;
  v16h f;
  *((float4*)&f)     = *(const float4*)(base);       // k = kb+0 .. kb+7
  *((float4*)&f + 1) = *(const float4*)(base + 16);  // k = kb+16 .. kb+23
  return f;
}

// C/D layout: lanes 0-15 -> N=lane, M=r; lanes 16-31 -> N=lane-16, M=r+8.
static __device__ inline void store_tile_f16(_Float16* __restrict__ p, int ld,
                                             int r0, int c0, v8f c) {
  const int lane = threadIdx.x & 31;
  const int n = lane & 15, mb = (lane >> 4) << 3;
#pragma unroll
  for (int r = 0; r < 8; ++r)
    p[(size_t)(r0 + mb + r) * ld + c0 + n] = (_Float16)c[r];
}

// LayerNorm over 64 rows x 256 cols: 4 lanes per row, shfl-xor 1/2 reduce.
// Row data cached in registers (16 x b128 LDS loads), packed b64 f16 stores.
static __device__ inline void layernorm_64x256(const float* __restrict__ xs,
                                               _Float16* __restrict__ dst,
                                               const float* __restrict__ w,
                                               const float* __restrict__ bvec) {
  const int tid = threadIdx.x;
  const int row = (tid >> 5) * 8 + ((tid & 31) >> 2);
  const int c0  = (tid & 3) * 64;
  const float4* xr4 = (const float4*)(xs + row * 256 + c0);
  float4 c[16];
  float s = 0.f;
#pragma unroll
  for (int i = 0; i < 16; ++i) {
    c[i] = xr4[i];
    s += c[i].x + c[i].y + c[i].z + c[i].w;
  }
  s += __shfl_xor(s, 1); s += __shfl_xor(s, 2);
  const float mu = s * (1.0f / 256.0f);
  float vs = 0.f;
#pragma unroll
  for (int i = 0; i < 16; ++i) {
    float dx = c[i].x - mu, dy = c[i].y - mu, dz = c[i].z - mu, dw = c[i].w - mu;
    vs += dx * dx + dy * dy + dz * dz + dw * dw;
  }
  vs += __shfl_xor(vs, 1); vs += __shfl_xor(vs, 2);
  const float rstd = rsqrtf(vs * (1.0f / 256.0f) + 1e-5f);
  uint2* d2 = (uint2*)(dst + row * 256 + c0);
#pragma unroll
  for (int i = 0; i < 16; ++i) {
    const float4 wv = *(const float4*)(w + c0 + i * 4);
    const float4 bv = *(const float4*)(bvec + c0 + i * 4);
    union { _Float16 h[4]; uint2 u; } o;
    o.h[0] = (_Float16)((c[i].x - mu) * rstd * wv.x + bv.x);
    o.h[1] = (_Float16)((c[i].y - mu) * rstd * wv.y + bv.y);
    o.h[2] = (_Float16)((c[i].z - mu) * rstd * wv.z + bv.z);
    o.h[3] = (_Float16)((c[i].w - mu) * rstd * wv.w + bv.w);
    d2[i] = o.u;
  }
}

// ---- prep: fp32 weights -> f16 in ws; expand rel-pos bias to [H][64][64] ----
__global__ void swin_prep(const float* __restrict__ qkv_w, const float* __restrict__ proj_w,
                          const float* __restrict__ fc1_w, const float* __restrict__ fc2_w,
                          const float* __restrict__ bias_table,
                          const int* __restrict__ rel_index, char* __restrict__ ws) {
  _Float16* wq = (_Float16*)(ws + WQKV_OFF);
  _Float16* wp = (_Float16*)(ws + WPROJ_OFF);
  _Float16* w1 = (_Float16*)(ws + WFC1_OFF);
  _Float16* w2 = (_Float16*)(ws + WFC2_OFF);
  float*    bf = (float*)(ws + BIAS_OFF);
  const int stride = gridDim.x * blockDim.x;
  const int t0 = blockIdx.x * blockDim.x + threadIdx.x;
  for (int i = t0; i < 768 * 256;  i += stride) wq[i] = (_Float16)qkv_w[i];
  for (int i = t0; i < 256 * 256;  i += stride) wp[i] = (_Float16)proj_w[i];
  for (int i = t0; i < 1024 * 256; i += stride) w1[i] = (_Float16)fc1_w[i];
  for (int i = t0; i < 256 * 1024; i += stride) w2[i] = (_Float16)fc2_w[i];
  for (int i = t0; i < HEADS * 64 * 64; i += stride) {
    int h = i >> 12, mn = i & 4095;
    bf[i] = bias_table[rel_index[mn] * HEADS + h];
  }
}

// ---- fused Swin block: one workgroup per window, 8 waves ----
__global__ __launch_bounds__(256)
void swin_block(const float* __restrict__ x,
                const float* __restrict__ qkv_b, const float* __restrict__ proj_b,
                const float* __restrict__ n1w, const float* __restrict__ n1b,
                const float* __restrict__ n2w, const float* __restrict__ n2b,
                const float* __restrict__ fc1b, const float* __restrict__ fc2b,
                const char* __restrict__ ws, float* __restrict__ out) {
  extern __shared__ char smem[];
  float*    xs   = (float*)(smem + XS_OFF);
  _Float16* hA   = (_Float16*)(smem + HA_OFF);    // ln1 out / attn out / ln2 out
  _Float16* qkvb = (_Float16*)(smem + QKV_OFF);   // qkv, later mlp hidden
  _Float16* pbuf = (_Float16*)(smem + PBUF_OFF);
  _Float16* vt   = (_Float16*)(smem + VT_OFF);    // V^T: [256 v-cols][64 tokens]

  const _Float16* wq = (const _Float16*)(ws + WQKV_OFF);
  const _Float16* wp = (const _Float16*)(ws + WPROJ_OFF);
  const _Float16* w1 = (const _Float16*)(ws + WFC1_OFF);
  const _Float16* w2 = (const _Float16*)(ws + WFC2_OFF);
  const float* biasf = (const float*)(ws + BIAS_OFF);

  const int tid  = threadIdx.x;
  const int wave = tid >> 5, lane = tid & 31;
  const int nn = lane & 15, mbase = (lane >> 4) << 3;
  const float* xg = x + (size_t)blockIdx.x * (TOKENS * DIM);
  const v8f vzero = {};

  // stage 0: residual into LDS via async direct-to-LDS path (ASYNCcnt-tracked)
#if __has_builtin(__builtin_amdgcn_global_load_async_to_lds_b128)
  for (int i = tid; i < TOKENS * DIM / 4; i += 256)
    __builtin_amdgcn_global_load_async_to_lds_b128(
        (glob_v4i)(xg + i * 4), (lds_v4i)(smem + XS_OFF + i * 16), 0, 0);
#if __has_builtin(__builtin_amdgcn_s_wait_asynccnt)
  __builtin_amdgcn_s_wait_asynccnt(0);
#else
  asm volatile("s_wait_asynccnt 0x0" ::: "memory");
#endif
#else
  for (int i = tid; i < TOKENS * DIM / 4; i += 256)
    ((float4*)xs)[i] = ((const float4*)xg)[i];
#endif
  __syncthreads();

  // stage 1: LN1 -> hA (f16)
  layernorm_64x256(xs, hA, n1w, n1b);
  __syncthreads();

  // stage 2: QKV = hA @ Wqkv^T + b   (4 x 48 tiles, K = 256)
  //          V tiles (ot >= 32) are additionally stored transposed into vt.
  for (int t = wave; t < 4 * 48; t += 8) {
    const int mt = t & 3, ot = t >> 2;
    v8f acc = {};
#pragma unroll
    for (int kt = 0; kt < 8; ++kt)
      acc = wmma32(load_frag_row(hA, DIM, mt * 16, kt * 32),
                   load_frag_row(wq, DIM, ot * 16, kt * 32), acc);
    const float bb = qkv_b[ot * 16 + nn];
#pragma unroll
    for (int r = 0; r < 8; ++r) acc[r] += bb;
    store_tile_f16(qkvb, 3 * DIM, mt * 16, ot * 16, acc);
    if (ot >= 32) {  // V region: vt[c][m] = V[m][c]
      _Float16* vtc = vt + (size_t)((ot - 32) * 16 + nn) * 64 + mt * 16 + mbase;
#pragma unroll
      for (int r = 0; r < 8; ++r) vtc[r] = (_Float16)acc[r];
    }
  }
  __syncthreads();

  // stage 3: attention, one head per wave. HEAD_DIM=32 -> 1 WMMA per S tile.
  {
    const int h = wave;
    const _Float16* qm = qkvb + h * HD;
    const _Float16* km = qkvb + DIM + h * HD;
    const _Float16* vth = vt + (size_t)(h * HD) * 64;   // [32 d][64 tokens]
    const float* bh = biasf + h * 64 * 64;
    _Float16* ph = pbuf + h * 64 * 64;
    const float scale = 0.17677669529663687f;  // 1/sqrt(32)

    v16h kfrag[4];
#pragma unroll
    for (int nt = 0; nt < 4; ++nt)
      kfrag[nt] = load_frag_row(km, 3 * DIM, nt * 16, 0);  // B = K^T via row loader

#pragma unroll
    for (int mt = 0; mt < 4; ++mt) {
      v16h a = load_frag_row(qm, 3 * DIM, mt * 16, 0);
      v8f srow[4];
#pragma unroll
      for (int nt = 0; nt < 4; ++nt) srow[nt] = wmma32(a, kfrag[nt], vzero);
      // softmax over the 64-wide row (bias added); rows live in 16-lane halves
#pragma unroll
      for (int r = 0; r < 8; ++r) {
        const int m = mt * 16 + mbase + r;
        float v[4], rmax = -3.0e38f;
#pragma unroll
        for (int nt = 0; nt < 4; ++nt) {
          v[nt] = srow[nt][r] * scale + bh[m * 64 + nt * 16 + nn];
          rmax = fmaxf(rmax, v[nt]);
        }
#pragma unroll
        for (int off = 8; off; off >>= 1) rmax = fmaxf(rmax, __shfl_xor(rmax, off));
        float rsum = 0.f;
#pragma unroll
        for (int nt = 0; nt < 4; ++nt) { v[nt] = __expf(v[nt] - rmax); rsum += v[nt]; }
#pragma unroll
        for (int off = 8; off; off >>= 1) rsum += __shfl_xor(rsum, off);
        const float inv = 1.0f / rsum;
#pragma unroll
        for (int nt = 0; nt < 4; ++nt)
          ph[m * 64 + nt * 16 + nn] = (_Float16)(v[nt] * inv);
      }
    }
    // O = P @ V  (4 x 2 tiles, K = 64); B from vt via the fast row loader
#pragma unroll
    for (int mt = 0; mt < 4; ++mt)
#pragma unroll
      for (int dt = 0; dt < 2; ++dt) {
        v8f acc = {};
#pragma unroll
        for (int kt = 0; kt < 2; ++kt)
          acc = wmma32(load_frag_row(ph, 64, mt * 16, kt * 32),
                       load_frag_row(vth, 64, dt * 16, kt * 32), acc);
        store_tile_f16(hA, DIM, mt * 16, h * HD + dt * 16, acc);
      }
  }
  __syncthreads();

  // stage 4: x += attn_out @ Wproj^T + b   (4 x 16 tiles, K = 256)
  for (int t = wave; t < 4 * 16; t += 8) {
    const int mt = t & 3, ot = t >> 2;
    v8f acc = {};
#pragma unroll
    for (int kt = 0; kt < 8; ++kt)
      acc = wmma32(load_frag_row(hA, DIM, mt * 16, kt * 32),
                   load_frag_row(wp, DIM, ot * 16, kt * 32), acc);
    const float bb = proj_b[ot * 16 + nn];
#pragma unroll
    for (int r = 0; r < 8; ++r)
      xs[(mt * 16 + mbase + r) * DIM + ot * 16 + nn] += acc[r] + bb;
  }
  __syncthreads();

  // stage 5: LN2 -> hA
  layernorm_64x256(xs, hA, n2w, n2b);
  __syncthreads();

  // stage 6: hidden = gelu(hA @ W1^T + b1)  (4 x 64 tiles, K = 256) -> qkvb region
  _Float16* g = qkvb;  // 64 x 1024 f16
  for (int t = wave; t < 4 * 64; t += 8) {
    const int mt = t & 3, ot = t >> 2;
    v8f acc = {};
#pragma unroll
    for (int kt = 0; kt < 8; ++kt)
      acc = wmma32(load_frag_row(hA, DIM, mt * 16, kt * 32),
                   load_frag_row(w1, DIM, ot * 16, kt * 32), acc);
    const float bb = fc1b[ot * 16 + nn];
#pragma unroll
    for (int r = 0; r < 8; ++r) {
      const float v = acc[r] + bb;
      const float ge = 0.5f * v * (1.0f + erff(v * 0.70710678118654752f));  // exact GELU
      g[(mt * 16 + mbase + r) * FF + ot * 16 + nn] = (_Float16)ge;
    }
  }
  __syncthreads();

  // stage 7: out = x + hidden @ W2^T + b2  (4 x 16 tiles, K = 1024)
  float* og = out + (size_t)blockIdx.x * (TOKENS * DIM);
  for (int t = wave; t < 4 * 16; t += 8) {
    const int mt = t & 3, ot = t >> 2;
    v8f acc = {};
#pragma unroll
    for (int kt = 0; kt < 32; ++kt)
      acc = wmma32(load_frag_row(g, FF, mt * 16, kt * 32),
                   load_frag_row(w2, FF, ot * 16, kt * 32), acc);
    const float bb = fc2b[ot * 16 + nn];
#pragma unroll
    for (int r = 0; r < 8; ++r) {
      const int row = mt * 16 + mbase + r, col = ot * 16 + nn;
      og[row * DIM + col] = xs[row * DIM + col] + acc[r] + bb;
    }
  }
}

extern "C" void kernel_launch(void* const* d_in, const int* in_sizes, int n_in,
                              void* d_out, int out_size, void* d_ws, size_t ws_size,
                              hipStream_t stream) {
  const float* x          = (const float*)d_in[0];
  const float* qkv_w      = (const float*)d_in[1];
  const float* qkv_b      = (const float*)d_in[2];
  const float* proj_w     = (const float*)d_in[3];
  const float* proj_b     = (const float*)d_in[4];
  const float* bias_table = (const float*)d_in[5];
  const float* n1w        = (const float*)d_in[6];
  const float* n1b        = (const float*)d_in[7];
  const float* n2w        = (const float*)d_in[8];
  const float* n2b        = (const float*)d_in[9];
  const float* fc1_w      = (const float*)d_in[10];
  const float* fc1_b      = (const float*)d_in[11];
  const float* fc2_w      = (const float*)d_in[12];
  const float* fc2_b      = (const float*)d_in[13];
  const int*   rel_index  = (const int*)d_in[14];
  float* out = (float*)d_out;
  char*  ws  = (char*)d_ws;

  const int B = in_sizes[0] / (TOKENS * DIM);  // 2048 windows

  swin_prep<<<512, 256, 0, stream>>>(qkv_w, proj_w, fc1_w, fc2_w, bias_table,
                                     rel_index, ws);

  (void)hipFuncSetAttribute((const void*)swin_block,
                            hipFuncAttributeMaxDynamicSharedMemorySize, LDS_BYTES);
  swin_block<<<dim3(B), dim3(256), LDS_BYTES, stream>>>(
      x, qkv_b, proj_b, n1w, n1b, n2w, n2b, fc1_b, fc2_b, ws, out);
}
